// MoE_3659312136194
// MI455X (gfx1250) — matrix operand order repaired and checked by reading, not compiled, not run on hardware
//
#include <hip/hip_runtime.h>
#include <hip/hip_bf16.h>

// ---------- types ----------
typedef __attribute__((ext_vector_type(16))) __bf16 bf16x16;
typedef __attribute__((ext_vector_type(8)))  __bf16 bf16x8;
typedef __attribute__((ext_vector_type(2)))  __bf16 bf16x2;
typedef __attribute__((ext_vector_type(8)))  float  f32x8;

union FragBF {
    bf16x16 v;
    bf16x8  h[2];
};

// f32 -> bf16 via native cast (clang legalizes fptrunc inline; on gfx1250 the
// packed pair gives the backend a shot at v_cvt_pk_bf16_f32).
__device__ __forceinline__ unsigned short f32_to_bf16(float f) {
    __bf16 b = (__bf16)f;
    unsigned short u;
    __builtin_memcpy(&u, &b, 2);
    return u;
}
__device__ __forceinline__ void store_bf16x2(unsigned short* dst, float a, float b) {
    bf16x2 p;
    p[0] = (__bf16)a;
    p[1] = (__bf16)b;
    __builtin_memcpy(dst, &p, 4);
}
__device__ __forceinline__ void store_bf16x4(unsigned short* dst, float4 v) {
    store_bf16x2(dst + 0, v.x, v.y);
    store_bf16x2(dst + 2, v.z, v.w);
}

__device__ __forceinline__ float fast_sigmoid(float x) {
    return __builtin_amdgcn_rcpf(1.0f + __expf(-x));   // TRANS ops, no IEEE div
}
__device__ __forceinline__ float silu_f(float x) {
    return x * fast_sigmoid(x);
}
__device__ __forceinline__ f32x8 zero8() {
    f32x8 z;
#pragma unroll
    for (int i = 0; i < 8; ++i) z[i] = 0.0f;
    return z;
}

// Tile geometry: block = 64(M) x 128(N), 8 waves (2x4), wave = 32x32 (2x2 WMMA), KT = 32
#define LDP 48   // LDS row pitch in ushorts (96B: 16B-aligned rows, conflict-spread)

// ============================================================================
// Fused SwiGLU up-projection. Pipeline per K-step:
//   fetch(k+1)->regs | ds_load frags(k) | 8x WMMA | convert+commit(k+1)->LDS | barrier
// so the global-load wait lands AFTER the WMMA block.
// ============================================================================
template <bool HAS_COMBINE>
__global__ __launch_bounds__(256) void moe_up_swiglu(
    const float* __restrict__ X,           // (T, D) fp32
    const float* __restrict__ W1b,         // gate weights base
    const float* __restrict__ W2b,         // up weights base
    const float* __restrict__ combine,     // (T, 8) or nullptr
    unsigned short* __restrict__ act,      // (T, actStride) bf16
    int D, int actStride, int nPerExpert,
    long long wExpStride)
{
    const int e      = blockIdx.z;
    const float* W1  = W1b + (long long)e * wExpStride;
    const float* W2  = W2b + (long long)e * wExpStride;
    const int blockM = blockIdx.y * 64;
    const int blockN = blockIdx.x * 128;
    const int tid    = threadIdx.x;
    const int lane   = tid & 31;
    const int wave   = tid >> 5;
    const int waveM  = wave >> 2;          // 0..1
    const int waveN  = wave & 3;           // 0..3
    const int half   = lane >> 4;          // 0/1
    const int lr     = lane & 15;

    __shared__ unsigned short lA [2][64  * LDP];
    __shared__ unsigned short lB1[2][128 * LDP];
    __shared__ unsigned short lB2[2][128 * LDP];

    struct UpRegs { float4 a[2]; float4 b1[4]; float4 b2[4]; };

    auto fetch = [&](int k0, UpRegs& s) {   // global -> regs (no LDS, no wait forced)
#pragma unroll
        for (int it = 0; it < 2; ++it) {
            int i  = tid + it * 256;
            int r  = i >> 3;
            int c4 = (i & 7) * 4;
            s.a[it] = *(const float4*)(X + (long long)(blockM + r) * D + k0 + c4);
        }
#pragma unroll
        for (int it = 0; it < 4; ++it) {
            int i  = tid + it * 256;
            int r  = i >> 3;
            int c4 = (i & 7) * 4;
            long long off = (long long)(blockN + r) * D + k0 + c4;
            s.b1[it] = *(const float4*)(W1 + off);
            s.b2[it] = *(const float4*)(W2 + off);
        }
    };
    auto commit = [&](const UpRegs& s, int b) {   // regs -> bf16 -> LDS
#pragma unroll
        for (int it = 0; it < 2; ++it) {
            int i  = tid + it * 256;
            int r  = i >> 3;
            int c4 = (i & 7) * 4;
            store_bf16x4(&lA[b][r * LDP + c4], s.a[it]);
        }
#pragma unroll
        for (int it = 0; it < 4; ++it) {
            int i  = tid + it * 256;
            int r  = i >> 3;
            int c4 = (i & 7) * 4;
            store_bf16x4(&lB1[b][r * LDP + c4], s.b1[it]);
            store_bf16x4(&lB2[b][r * LDP + c4], s.b2[it]);
        }
    };

    f32x8 accG[2][2], accU[2][2];
#pragma unroll
    for (int i = 0; i < 2; ++i)
#pragma unroll
        for (int j = 0; j < 2; ++j) { accG[i][j] = zero8(); accU[i][j] = zero8(); }

    UpRegs s;
    fetch(0, s);
    commit(s, 0);
    __syncthreads();

    int buf = 0;
    for (int k0 = 0; k0 < D; k0 += 32) {
        const bool more = (k0 + 32 < D);
        if (more) fetch(k0 + 32, s);                     // issue loads early

        FragBF a[2], b1[2], b2[2];
#pragma unroll
        for (int mi = 0; mi < 2; ++mi) {
            int row = waveM * 32 + mi * 16 + lr;
            const unsigned short* base = &lA[buf][row * LDP];
            a[mi].h[0] = *(const bf16x8*)(base + half * 16);       // K 0..7 / 16..23
            a[mi].h[1] = *(const bf16x8*)(base + half * 16 + 8);   // K 8..15 / 24..31
        }
#pragma unroll
        for (int ni = 0; ni < 2; ++ni) {
            int col = waveN * 32 + ni * 16 + lr;
            const unsigned short* p1 = &lB1[buf][col * LDP + half * 16]; // K 0..15/16..31
            b1[ni].h[0] = *(const bf16x8*)(p1);
            b1[ni].h[1] = *(const bf16x8*)(p1 + 8);
            const unsigned short* p2 = &lB2[buf][col * LDP + half * 16];
            b2[ni].h[0] = *(const bf16x8*)(p2);
            b2[ni].h[1] = *(const bf16x8*)(p2 + 8);
        }
#pragma unroll
        for (int mi = 0; mi < 2; ++mi)
#pragma unroll
            for (int ni = 0; ni < 2; ++ni) {
                accG[mi][ni] = __builtin_amdgcn_wmma_f32_16x16x32_bf16(
                    false, a[mi].v, false, b1[ni].v, (short)0, accG[mi][ni], false, false);
                accU[mi][ni] = __builtin_amdgcn_wmma_f32_16x16x32_bf16(
                    false, a[mi].v, false, b2[ni].v, (short)0, accU[mi][ni], false, false);
            }
        if (more) commit(s, buf ^ 1);                    // loadcnt wait lands here
        __syncthreads();
        buf ^= 1;
    }

    // ---- epilogue: silu(g)*u (* combine), store bf16 ----
    float cs[2][8];
#pragma unroll
    for (int mi = 0; mi < 2; ++mi)
#pragma unroll
        for (int i = 0; i < 8; ++i) {
            if (HAS_COMBINE) {
                int t = blockM + waveM * 32 + mi * 16 + half * 8 + i;
                cs[mi][i] = combine[t * 8 + e];
            } else {
                cs[mi][i] = 1.0f;
            }
        }
#pragma unroll
    for (int mi = 0; mi < 2; ++mi) {
#pragma unroll
        for (int ni = 0; ni < 2; ++ni) {
            int col = blockN + waveN * 32 + ni * 16 + lr;
#pragma unroll
            for (int i = 0; i < 8; ++i) {
                int t = blockM + waveM * 32 + mi * 16 + half * 8 + i;
                float g = accG[mi][ni][i];
                float u = accU[mi][ni][i];
                float v = silu_f(g) * u * cs[mi][i];
                act[(long long)t * actStride + (long long)e * nPerExpert + col] = f32_to_bf16(v);
            }
        }
    }
}

// ============================================================================
// Down-projection, same reg-prefetch pipeline:
//   out[t,d] (=|+=) sum_k act_bf16[t,k] * W[k/Hper, d, k%Hper]
// ============================================================================
template <bool ACCUM>
__global__ __launch_bounds__(256) void moe_down(
    const unsigned short* __restrict__ act,  // (T, K) bf16
    const float* __restrict__ W,
    float* __restrict__ out,                 // (T, D) fp32
    int D, int K, int Hper)
{
    const int blockM = blockIdx.y * 64;
    const int blockN = blockIdx.x * 128;
    const int tid    = threadIdx.x;
    const int lane   = tid & 31;
    const int wave   = tid >> 5;
    const int waveM  = wave >> 2;
    const int waveN  = wave & 3;
    const int half   = lane >> 4;
    const int lr     = lane & 15;

    __shared__ unsigned short lA[2][64  * LDP];
    __shared__ unsigned short lB[2][128 * LDP];

    struct DnRegs { unsigned long long a[2]; float4 b[4]; };

    auto fetch = [&](int k0, DnRegs& s) {
        const int ee   = k0 / Hper;              // expert slab (KT divides Hper)
        const int koff = k0 % Hper;
#pragma unroll
        for (int it = 0; it < 2; ++it) {         // A: 64x32 bf16, raw 8B chunks
            int i  = tid + it * 256;
            int r  = i >> 3;
            int c4 = (i & 7) * 4;
            s.a[it] = *(const unsigned long long*)(act + (long long)(blockM + r) * K + k0 + c4);
        }
#pragma unroll
        for (int it = 0; it < 4; ++it) {         // W: 128x32 fp32
            int i  = tid + it * 256;
            int r  = i >> 3;
            int c4 = (i & 7) * 4;
            s.b[it] = *(const float4*)(W + (long long)ee * D * Hper
                                         + (long long)(blockN + r) * Hper + koff + c4);
        }
    };
    auto commit = [&](const DnRegs& s, int b) {
#pragma unroll
        for (int it = 0; it < 2; ++it) {
            int i  = tid + it * 256;
            int r  = i >> 3;
            int c4 = (i & 7) * 4;
            *(unsigned long long*)&lA[b][r * LDP + c4] = s.a[it];
        }
#pragma unroll
        for (int it = 0; it < 4; ++it) {
            int i  = tid + it * 256;
            int r  = i >> 3;
            int c4 = (i & 7) * 4;
            store_bf16x4(&lB[b][r * LDP + c4], s.b[it]);
        }
    };

    f32x8 acc[2][2];
#pragma unroll
    for (int i = 0; i < 2; ++i)
#pragma unroll
        for (int j = 0; j < 2; ++j) acc[i][j] = zero8();

    DnRegs s;
    fetch(0, s);
    commit(s, 0);
    __syncthreads();

    int buf = 0;
    for (int k0 = 0; k0 < K; k0 += 32) {
        const bool more = (k0 + 32 < K);
        if (more) fetch(k0 + 32, s);

        FragBF a[2], b[2];
#pragma unroll
        for (int mi = 0; mi < 2; ++mi) {
            int row = waveM * 32 + mi * 16 + lr;
            const unsigned short* base = &lA[buf][row * LDP];
            a[mi].h[0] = *(const bf16x8*)(base + half * 16);
            a[mi].h[1] = *(const bf16x8*)(base + half * 16 + 8);
        }
#pragma unroll
        for (int ni = 0; ni < 2; ++ni) {
            int col = waveN * 32 + ni * 16 + lr;
            const unsigned short* p = &lB[buf][col * LDP + half * 16];
            b[ni].h[0] = *(const bf16x8*)(p);
            b[ni].h[1] = *(const bf16x8*)(p + 8);
        }
#pragma unroll
        for (int mi = 0; mi < 2; ++mi)
#pragma unroll
            for (int ni = 0; ni < 2; ++ni)
                acc[mi][ni] = __builtin_amdgcn_wmma_f32_16x16x32_bf16(
                    false, a[mi].v, false, b[ni].v, (short)0, acc[mi][ni], false, false);
        if (more) commit(s, buf ^ 1);
        __syncthreads();
        buf ^= 1;
    }

#pragma unroll
    for (int mi = 0; mi < 2; ++mi) {
#pragma unroll
        for (int ni = 0; ni < 2; ++ni) {
            int d = blockN + waveN * 32 + ni * 16 + lr;
#pragma unroll
            for (int i = 0; i < 8; ++i) {
                int t = blockM + waveM * 32 + mi * 16 + half * 8 + i;
                long long idx = (long long)t * D + d;
                float v = acc[mi][ni][i];
                if (ACCUM) out[idx] += v;
                else       out[idx]  = v;
            }
        }
    }
}

// ============================================================================
// Router: sigmoid(x @ rw^T), biased top-2, normalized combine weights (T,8).
// ============================================================================
__global__ __launch_bounds__(256) void moe_router(
    const float* __restrict__ X,
    const float* __restrict__ RW,        // (8, D)
    const float* __restrict__ bias,      // (8)
    float* __restrict__ combine,         // (T, 8)
    int D)
{
    const int wave = threadIdx.x >> 5;
    const int lane = threadIdx.x & 31;
    const int t    = blockIdx.x * 8 + wave;

    float acc[8];
#pragma unroll
    for (int e = 0; e < 8; ++e) acc[e] = 0.0f;

    for (int k = lane; k < D; k += 32) {
        float xv = X[(long long)t * D + k];
#pragma unroll
        for (int e = 0; e < 8; ++e) acc[e] += xv * RW[(long long)e * D + k];
    }
#pragma unroll
    for (int e = 0; e < 8; ++e)
#pragma unroll
        for (int off = 16; off > 0; off >>= 1)
            acc[e] += __shfl_xor(acc[e], off, 32);

    if (lane == 0) {
        float sc[8], sel[8];
#pragma unroll
        for (int e = 0; e < 8; ++e) {
            sc[e]  = fast_sigmoid(acc[e]);
            sel[e] = sc[e] + bias[e];
        }
        int i1 = 0;
#pragma unroll
        for (int e = 1; e < 8; ++e) if (sel[e] > sel[i1]) i1 = e;
        int i2 = (i1 == 0) ? 1 : 0;
#pragma unroll
        for (int e = 0; e < 8; ++e) if (e != i1 && sel[e] > sel[i2]) i2 = e;
        float s1 = sc[i1], s2 = sc[i2];
        float inv = __builtin_amdgcn_rcpf(s1 + s2 + 1e-9f);
#pragma unroll
        for (int e = 0; e < 8; ++e) combine[t * 8 + e] = 0.0f;
        combine[t * 8 + i1] = s1 * inv;
        combine[t * 8 + i2] = s2 * inv;
    }
}

// ============================================================================
extern "C" void kernel_launch(void* const* d_in, const int* in_sizes, int n_in,
                              void* d_out, int out_size, void* d_ws, size_t ws_size,
                              hipStream_t stream) {
    const float* x           = (const float*)d_in[0];
    const float* w12         = (const float*)d_in[1];
    const float* w3          = (const float*)d_in[2];
    const float* router_w    = (const float*)d_in[3];
    const float* expert_bias = (const float*)d_in[4];
    const float* sw1         = (const float*)d_in[5];
    const float* sw2         = (const float*)d_in[6];
    const float* sw3         = (const float*)d_in[7];
    float* out = (float*)d_out;

    const int T = 4096, D = 2048, E = 8, Hs = 8192, Hr = 1024;

    float* combine      = (float*)d_ws;                                   // 128 KB
    unsigned short* act = (unsigned short*)((char*)d_ws + (1 << 20));     // 64 MB, reused

    dim3 blk(256);

    // 1) shared expert up: act(T,8192) = silu(x@sw1^T) * (x@sw2^T)   [bf16]
    moe_up_swiglu<false><<<dim3(Hs / 128, T / 64, 1), blk, 0, stream>>>(
        x, sw1, sw2, nullptr, act, D, Hs, Hs, 0LL);

    // 2) shared expert down: out = act @ sw3^T    (write)
    moe_down<false><<<dim3(D / 128, T / 64, 1), blk, 0, stream>>>(
        act, sw3, out, D, Hs, Hs);

    // 3) router -> combine (T,8)
    moe_router<<<dim3(T / 8), blk, 0, stream>>>(x, router_w, expert_bias, combine, D);

    // 4) routed up (grid.z = expert), scaled by combine[t,e]:
    //    act(T, e*1024 + h) = combine * silu(x@w1_e^T) * (x@w2_e^T)
    moe_up_swiglu<true><<<dim3(Hr / 128, T / 64, E), blk, 0, stream>>>(
        x, w12, w12 + (long long)Hr * D, combine, act, D, E * Hr, Hr,
        (long long)2 * Hr * D);

    // 5) routed down (grouped over (e,h), K = 8192, Hper = 1024): out += ...
    moe_down<true><<<dim3(D / 128, T / 64, 1), blk, 0, stream>>>(
        act, w3, out, D, E * Hr, Hr);
}
